// RwkvSelfAttention_8658654069008
// MI455X (gfx1250) — compile-verified, matrix operand order checked by reading, not run on hardware
//
#include <hip/hip_runtime.h>

#define Bb 8
#define Tt 1024
#define Cc 2048
#define Mm (Bb * Tt)   // 8192 rows
#define BK 32

typedef __attribute__((ext_vector_type(16))) __bf16 v16bf;
typedef __attribute__((ext_vector_type(8)))  float  v8f;

__device__ __forceinline__ unsigned short f32_bf16(float f) {
    unsigned int u = __float_as_uint(f);
    u += 0x7FFFu + ((u >> 16) & 1u);   // round-to-nearest-even
    return (unsigned short)(u >> 16);
}

__device__ __forceinline__ unsigned int pack_bf16x2(float a, float b) {
    return (unsigned int)f32_bf16(a) | ((unsigned int)f32_bf16(b) << 16);
}

// CDNA5 async global -> LDS mover (tracked by ASYNCcnt, no VGPR staging).
__device__ __forceinline__ void async_load_b128(unsigned lds_off, unsigned long long gaddr) {
    asm volatile("global_load_async_to_lds_b128 %0, %1, off"
                 :: "v"(lds_off), "v"(gaddr)
                 : "memory");
}
__device__ __forceinline__ void wait_async0() {
    asm volatile("s_wait_asynccnt 0x0" ::: "memory");
}

// Generic->LDS pointers keep the raw LDS byte offset in the low 32 bits
// (shared aperture lives in bits [63:32], ISA 10.2).
__device__ __forceinline__ unsigned lds_offset(const void* p) {
    return (unsigned)(unsigned long long)p;
}

// ---------------------------------------------------------------------------
// Kernel 1: time-shift mix  ->  xk/xv/xr in bf16.  4 elems/thread (same row).
// ---------------------------------------------------------------------------
__global__ __launch_bounds__(256) void mix_kernel(
    const float* __restrict__ x,
    const float* __restrict__ mk, const float* __restrict__ mv,
    const float* __restrict__ mr,
    unsigned short* __restrict__ xk, unsigned short* __restrict__ xv,
    unsigned short* __restrict__ xr)
{
    size_t i4 = (size_t)blockIdx.x * blockDim.x + threadIdx.x;
    size_t i  = i4 * 4;
    int c = (int)(i % Cc);
    int t = (int)((i / Cc) % Tt);

    float4 xc = *(const float4*)(x + i);
    float4 xs = make_float4(0.f, 0.f, 0.f, 0.f);
    if (t != 0) xs = *(const float4*)(x + i - Cc);

    float4 k4 = *(const float4*)(mk + c);
    float4 v4 = *(const float4*)(mv + c);
    float4 r4 = *(const float4*)(mr + c);

    uint2 ok, ov, orr;
    ok.x  = pack_bf16x2(xc.x * k4.x + xs.x * (1.f - k4.x),
                        xc.y * k4.y + xs.y * (1.f - k4.y));
    ok.y  = pack_bf16x2(xc.z * k4.z + xs.z * (1.f - k4.z),
                        xc.w * k4.w + xs.w * (1.f - k4.w));
    ov.x  = pack_bf16x2(xc.x * v4.x + xs.x * (1.f - v4.x),
                        xc.y * v4.y + xs.y * (1.f - v4.y));
    ov.y  = pack_bf16x2(xc.z * v4.z + xs.z * (1.f - v4.z),
                        xc.w * v4.w + xs.w * (1.f - v4.w));
    orr.x = pack_bf16x2(xc.x * r4.x + xs.x * (1.f - r4.x),
                        xc.y * r4.y + xs.y * (1.f - r4.y));
    orr.y = pack_bf16x2(xc.z * r4.z + xs.z * (1.f - r4.z),
                        xc.w * r4.w + xs.w * (1.f - r4.w));

    *(uint2*)(xk + i) = ok;
    *(uint2*)(xv + i) = ov;
    *(uint2*)(xr + i) = orr;
}

// ---------------------------------------------------------------------------
// Kernel 2: fp32 -> bf16 weight conversion, 4 elems/thread.
// ---------------------------------------------------------------------------
__global__ __launch_bounds__(256) void cvtw_kernel(
    const float* __restrict__ w, unsigned short* __restrict__ o)
{
    size_t i = ((size_t)blockIdx.x * blockDim.x + threadIdx.x) * 4;
    float4 f = *(const float4*)(w + i);
    uint2 u;
    u.x = pack_bf16x2(f.x, f.y);
    u.y = pack_bf16x2(f.z, f.w);
    *(uint2*)(o + i) = u;
}

// ---------------------------------------------------------------------------
// Kernel 3: WMMA bf16 GEMM with double-buffered async global->LDS staging.
//   Out[m][n] = sum_k A[m][k] * W[n][k]      (W row-major [N][K] == einsum dc)
//   Block tile 128x128, BK=32, 8 waves (wave32), each wave 2x4 16x16 subtiles.
//   Running 64-bit pointers advance by BK*2=64 bytes/iteration (no per-iter
//   address recompute).
// ---------------------------------------------------------------------------
__global__ __launch_bounds__(256) void gemm_bf16(
    const unsigned short* __restrict__ A,   // [Mm][Cc] bf16
    const unsigned short* __restrict__ W,   // [Cc][Cc] bf16 (N-major)
    float* __restrict__ Out)                // [Mm][Cc] f32
{
    __shared__ __attribute__((aligned(16))) unsigned short Alds[2][128 * BK];
    __shared__ __attribute__((aligned(16))) unsigned short Blds[2][128 * BK];

    const int tid   = threadIdx.x;
    const int lane  = tid & 31;
    const int wave  = tid >> 5;
    const int r     = lane & 15;
    const int half  = lane >> 4;
    const int waveM = wave & 3;     // 4 row groups of 32
    const int waveN = wave >> 2;    // 2 col groups of 64
    const int bN0   = blockIdx.x * 128;
    const int bM0   = blockIdx.y * 128;

    // per-thread cooperative-load coordinates: 512 x b128 per tile, 2/thread
    const int ldRow0 = tid >> 2;                 // vi = tid
    const int ldCol0 = (tid & 3) << 3;
    const int ldRow1 = (tid + 256) >> 2;         // vi = tid + 256
    const int ldCol1 = ((tid + 256) & 3) << 3;

    const unsigned aLo[2] = { lds_offset(&Alds[0][0]), lds_offset(&Alds[1][0]) };
    const unsigned bLo[2] = { lds_offset(&Blds[0][0]), lds_offset(&Blds[1][0]) };
    const unsigned e0 = (unsigned)(ldRow0 * BK + ldCol0) * 2u;  // LDS byte offs
    const unsigned e1 = (unsigned)(ldRow1 * BK + ldCol1) * 2u;

    // running global pointers, advanced by BK bf16 = 64 bytes per K-step
    unsigned long long gA0 = (unsigned long long)&A[(size_t)(bM0 + ldRow0) * Cc + ldCol0];
    unsigned long long gW0 = (unsigned long long)&W[(size_t)(bN0 + ldRow0) * Cc + ldCol0];
    unsigned long long gA1 = (unsigned long long)&A[(size_t)(bM0 + ldRow1) * Cc + ldCol1];
    unsigned long long gW1 = (unsigned long long)&W[(size_t)(bN0 + ldRow1) * Cc + ldCol1];

    v8f acc[2][4] = {};

    // prologue: fill buffer 0 for kk = 0
    async_load_b128(aLo[0] + e0, gA0);
    async_load_b128(bLo[0] + e0, gW0);
    async_load_b128(aLo[0] + e1, gA1);
    async_load_b128(bLo[0] + e1, gW1);
    gA0 += 64; gW0 += 64; gA1 += 64; gW1 += 64;

    int p = 0;
    for (int kk = 0; kk < Cc; kk += BK) {
        wait_async0();        // own async loads for buf p landed in LDS
        __syncthreads();      // everyone's landed; everyone done reading buf 1-p

        if (kk + BK < Cc) {   // prefetch next K-step into the other buffer
            async_load_b128(aLo[1 - p] + e0, gA0);
            async_load_b128(bLo[1 - p] + e0, gW0);
            async_load_b128(aLo[1 - p] + e1, gA1);
            async_load_b128(bLo[1 - p] + e1, gW1);
            gA0 += 64; gW0 += 64; gA1 += 64; gW1 += 64;
        }

        union Frag { v16bf v; uint4 u[2]; };
        Frag fa[2], fb[4];
        #pragma unroll
        for (int mi = 0; mi < 2; ++mi) {
            int arow = waveM * 32 + mi * 16 + r;
            fa[mi].u[0] = *(const uint4*)&Alds[p][arow * BK + half * 8];
            fa[mi].u[1] = *(const uint4*)&Alds[p][arow * BK + 16 + half * 8];
        }
        #pragma unroll
        for (int ni = 0; ni < 4; ++ni) {
            int brow = waveN * 64 + ni * 16 + r;
            fb[ni].u[0] = *(const uint4*)&Blds[p][brow * BK + half * 8];
            fb[ni].u[1] = *(const uint4*)&Blds[p][brow * BK + 16 + half * 8];
        }
        #pragma unroll
        for (int mi = 0; mi < 2; ++mi)
            #pragma unroll
            for (int ni = 0; ni < 4; ++ni)
                acc[mi][ni] = __builtin_amdgcn_wmma_f32_16x16x32_bf16(
                    false, fa[mi].v, false, fb[ni].v,
                    (short)0, acc[mi][ni], false, false);

        p ^= 1;
    }

    // C/D layout: lane%16 -> N, VGPR i -> M=i (lanes 0-15) / M=8+i (lanes 16-31)
    #pragma unroll
    for (int mi = 0; mi < 2; ++mi) {
        #pragma unroll
        for (int ni = 0; ni < 4; ++ni) {
            int col  = bN0 + waveN * 64 + ni * 16 + r;
            int row0 = bM0 + waveM * 32 + mi * 16 + half * 8;
            #pragma unroll
            for (int i = 0; i < 8; ++i)
                Out[(size_t)(row0 + i) * Cc + col] = acc[mi][ni][i];
        }
    }
}

// ---------------------------------------------------------------------------
// Kernel 4: stabilized WKV scan, fused with sigmoid(r) * wkv -> bf16 y.
// One thread per (b, c) channel; coalesced stride-C accesses.
// ---------------------------------------------------------------------------
__global__ __launch_bounds__(256) void wkv_kernel(
    const float* __restrict__ kf, const float* __restrict__ vf,
    const float* __restrict__ rf,
    const float* __restrict__ td, const float* __restrict__ tf,
    unsigned short* __restrict__ y)
{
    int idx = blockIdx.x * blockDim.x + threadIdx.x;   // 0 .. B*C-1
    int b = idx / Cc;
    int c = idx - b * Cc;

    float w = -__expf(td[c]);
    float u = tf[c];
    float num = 0.f, den = 0.f, mx = -1e38f;
    size_t base = (size_t)b * Tt * Cc + c;

    for (int t = 0; t < Tt; ++t) {
        size_t o = base + (size_t)t * Cc;
        float kt = kf[o], vt = vf[o], rt = rf[o];

        float ku  = kt + u;
        float mo  = fmaxf(mx, ku);
        float e1  = __expf(mx - mo);
        float e2  = __expf(ku - mo);
        float out = (e1 * num + e2 * vt) / (e1 * den + e2);

        float mw  = mx + w;
        float ms  = fmaxf(mw, kt);
        float e1s = __expf(mw - ms);
        float e2s = __expf(kt - ms);
        num = e1s * num + e2s * vt;
        den = e1s * den + e2s;
        mx  = ms;

        float sr = 1.f / (1.f + __expf(-rt));
        y[o] = f32_bf16(sr * out);
    }
}

// ---------------------------------------------------------------------------
extern "C" void kernel_launch(void* const* d_in, const int* in_sizes, int n_in,
                              void* d_out, int out_size, void* d_ws, size_t ws_size,
                              hipStream_t stream) {
    (void)in_sizes; (void)n_in; (void)out_size; (void)ws_size;

    const float* x  = (const float*)d_in[0];
    const float* td = (const float*)d_in[1];
    const float* tf = (const float*)d_in[2];
    const float* mk = (const float*)d_in[3];
    const float* mv = (const float*)d_in[4];
    const float* mr = (const float*)d_in[5];
    const float* Wk = (const float*)d_in[6];
    const float* Wv = (const float*)d_in[7];
    const float* Wr = (const float*)d_in[8];
    const float* Wo = (const float*)d_in[9];
    float* out = (float*)d_out;

    const size_t nTok = (size_t)Mm * Cc;   // 16,777,216
    const size_t nW   = (size_t)Cc * Cc;   //  4,194,304

    char* ws = (char*)d_ws;
    unsigned short* xk  = (unsigned short*)ws; ws += nTok * 2;
    unsigned short* xv  = (unsigned short*)ws; ws += nTok * 2;
    unsigned short* xr  = (unsigned short*)ws; ws += nTok * 2;
    unsigned short* wkb = (unsigned short*)ws; ws += nW * 2;
    unsigned short* wvb = (unsigned short*)ws; ws += nW * 2;
    unsigned short* wrb = (unsigned short*)ws; ws += nW * 2;
    unsigned short* wob = (unsigned short*)ws; ws += nW * 2;
    float* kf = (float*)ws; ws += nTok * 4;
    float* vf = (float*)ws; ws += nTok * 4;
    float* rf = (float*)ws; ws += nTok * 4;
    unsigned short* y = (unsigned short*)ws; ws += nTok * 2;

    // 1) time-shift mixes -> bf16
    mix_kernel<<<(int)(nTok / 4 / 256), 256, 0, stream>>>(x, mk, mv, mr, xk, xv, xr);

    // 2) weights -> bf16
    int wBlocks = (int)(nW / 4 / 256);
    cvtw_kernel<<<wBlocks, 256, 0, stream>>>(Wk, wkb);
    cvtw_kernel<<<wBlocks, 256, 0, stream>>>(Wv, wvb);
    cvtw_kernel<<<wBlocks, 256, 0, stream>>>(Wr, wrb);
    cvtw_kernel<<<wBlocks, 256, 0, stream>>>(Wo, wob);

    // 3) projections: k, v, r
    dim3 gGrid(Cc / 128, Mm / 128);   // 16 x 64
    gemm_bf16<<<gGrid, 256, 0, stream>>>(xk, wkb, kf);
    gemm_bf16<<<gGrid, 256, 0, stream>>>(xv, wvb, vf);
    gemm_bf16<<<gGrid, 256, 0, stream>>>(xr, wrb, rf);

    // 4) WKV scan + sigmoid(r)*wkv -> bf16
    wkv_kernel<<<(Bb * Cc) / 256, 256, 0, stream>>>(kf, vf, rf, td, tf, y);

    // 5) output projection -> d_out (fp32)
    gemm_bf16<<<gGrid, 256, 0, stream>>>(y, wob, out);
}
